// SelfCorrelationPercPooling_31164282700122
// MI455X (gfx1250) — matrix operand order repaired and checked by reading, not compiled, not run on hardware
//
#include <hip/hip_runtime.h>

typedef __attribute__((ext_vector_type(16))) __bf16 v16bf;
typedef __attribute__((ext_vector_type(8)))  float  v8f;

#define BATCH   8
#define MDIM    4096     // H*W
#define CDIM    256      // feature dim
#define NPOOL   256
#define ROWS    16       // corr rows per workgroup
#define STRIDE  4104     // 4096 + 8 floats pad (breaks LDS bank alignment across rows)
#define THREADS 256

__global__ void __launch_bounds__(THREADS, 1)
selfcorr_percpool_kernel(const float* __restrict__ x, float* __restrict__ out) {
    extern __shared__ float corr[];   // [ROWS][STRIDE] = 262,656 bytes dynamic LDS

    const int tid  = threadIdx.x;
    const int wave = tid >> 5;        // 8 waves (wave32)
    const int lane = tid & 31;
    const int b    = blockIdx.x >> 8;     // batch
    const int tile = blockIdx.x & 255;    // row strip
    const int r0   = tile * ROWS;

    const int mrow = lane & 15;       // A-row (M) / B-col (N) index for this lane
    const int half = lane >> 4;       // lane half selects K sub-ranges per ISA layout

    const float* xb   = x + (size_t)b * MDIM * CDIM;
    const float* arow = xb + (size_t)(r0 + mrow) * CDIM;

    // ---- Phase 1: WMMA bf16 GEMM: corr[0:16][0:4096] = A * A^T / C -------------
    for (int ct = wave; ct < MDIM / 16; ct += 8) {       // wave-uniform loop
        const int colbase = ct * 16;
        const float* brow = xb + (size_t)(colbase + mrow) * CDIM;
        v8f acc = {};
        #pragma unroll
        for (int kk = 0; kk < CDIM / 32; ++kk) {
            const int base = kk * 32;
            // A fragment (16x32 bf16): lanes 0-15 K={0..7,16..23}, lanes 16-31 K={8..15,24..31}
            v16bf afrag;
            {
                const float* pa0 = arow + base + half * 8;
                const float* pa1 = arow + base + 16 + half * 8;
                #pragma unroll
                for (int e = 0; e < 8; ++e) {
                    afrag[e]     = (__bf16)pa0[e];
                    afrag[e + 8] = (__bf16)pa1[e];
                }
            }
            // B fragment (32x16 bf16): lanes 0-15 K={0..7,16..23}, lanes 16-31 K={8..15,24..31}
            v16bf bfrag;
            {
                const float* pb0 = brow + base + half * 8;
                const float* pb1 = brow + base + 16 + half * 8;
                #pragma unroll
                for (int e = 0; e < 8; ++e) {
                    bfrag[e]     = (__bf16)pb0[e];
                    bfrag[e + 8] = (__bf16)pb1[e];
                }
            }
            acc = __builtin_amdgcn_wmma_f32_16x16x32_bf16(
                false, afrag, false, bfrag, (short)0, acc, false, false);
        }
        // D layout: VGPR v, lane L -> row = v + 8*(L/16), col = colbase + (L%16)
        #pragma unroll
        for (int v = 0; v < 8; ++v) {
            corr[(v + 8 * half) * STRIDE + colbase + mrow] = acc[v] * (1.0f / CDIM);
        }
    }
    __syncthreads();

    // ---- Phase 2: bitonic sort each of the 16 rows, descending, in LDS ---------
    for (int k = 2; k <= MDIM; k <<= 1) {
        for (int j = k >> 1; j > 0; j >>= 1) {
            for (int t = tid; t < ROWS * (MDIM / 2); t += THREADS) {
                const int row = t >> 11;              // MDIM/2 = 2048 pairs per row
                const int p   = t & (MDIM / 2 - 1);
                const int i   = ((p & ~(j - 1)) << 1) | (p & (j - 1));
                float* r = corr + row * STRIDE;
                const float va = r[i];
                const float vb = r[i + j];
                const bool desc = ((i & k) == 0);     // descending overall
                const bool sw   = desc ? (va < vb) : (va > vb);
                if (sw) { r[i] = vb; r[i + j] = va; }
            }
            __syncthreads();
        }
    }

    // ---- Phase 3: gather percentile ranks (linspace(1, M-1, 256), rint) --------
    const float step = (float)(MDIM - 2) / (float)(NPOOL - 1);   // 4094/255
    for (int t = tid; t < ROWS * NPOOL; t += THREADS) {
        const int row  = t >> 8;
        const int r    = t & (NPOOL - 1);
        const int rank = (int)rintf(1.0f + step * (float)r);
        out[((size_t)(b * MDIM + r0 + row)) * NPOOL + r] = corr[row * STRIDE + rank];
    }
}

extern "C" void kernel_launch(void* const* d_in, const int* in_sizes, int n_in,
                              void* d_out, int out_size, void* d_ws, size_t ws_size,
                              hipStream_t stream) {
    const float* x = (const float*)d_in[0];
    float* out = (float*)d_out;
    const size_t shmem = (size_t)ROWS * STRIDE * sizeof(float);  // 262,656 B (<320 KB/WG)
    (void)hipFuncSetAttribute((const void*)selfcorr_percpool_kernel,
                              hipFuncAttributeMaxDynamicSharedMemorySize, (int)shmem);
    dim3 grid(BATCH * (MDIM / ROWS));   // 8 * 256 = 2048 workgroups
    selfcorr_percpool_kernel<<<grid, THREADS, shmem, stream>>>(x, out);
}
